// MPNN_43894565765750
// MI455X (gfx1250) — matrix operand order repaired and checked by reading, not compiled, not run on hardware
//
#include <hip/hip_runtime.h>
#include <math.h>

// ---------------- problem constants ----------------
#define NN   50000      // nodes
#define NE   800000     // edges
#define NG   500        // graphs
#define DIM  64
#define DEPTH 5

// ---------------- workspace layout (bytes) ----------------
#define OFF_X    ((size_t)0)                      // x     f32 [NN*64]   12.8 MB
#define OFF_AGG  (OFF_X   + (size_t)NN*64*4)      // agg   f32 [NN*64]   12.8 MB
#define OFF_XH   (OFF_AGG + (size_t)NN*64*4)      // xh    f16 [NN*64]    6.4 MB
#define OFF_RBF  (OFF_XH  + (size_t)NN*64*2)      // rbf   f16 [NE*16]   25.6 MB
#define OFF_POOL (OFF_RBF + (size_t)NE*16*2)      // pool  f32 [NG*64]
#define OFF_W    (OFF_POOL+ (size_t)NG*64*4)      // swizzled f16 weights

// per-layer swizzled weight block: 52 fragments of 512 f16 (contiguous)
//   [0,20)  mW1 (K padded 160, 5 ktiles x 4 ntiles)
//   [20,28) mW2 (K=64, 2x4)
//   [28,44) uW1 (K=128, 4x4)
//   [44,52) uW2 (K=64, 2x4)
#define FRAGS_PER_LAYER 52
#define WLAYER ((size_t)FRAGS_PER_LAYER*512)

#define EDGE_BLOCKS 1024
#define NODE_BLOCKS 512
#define NTILE_E (NE/16)     // 50000
#define NTILE_N (NN/16)     // 3125

typedef __attribute__((ext_vector_type(16))) _Float16 v16h;
typedef __attribute__((ext_vector_type(8)))  _Float16 v8h;
typedef __attribute__((ext_vector_type(8)))  float    v8f;

#define TS 72   // LDS transpose-tile row stride in f16 (16B aligned, staggers banks)

// ---- gfx1250 async global->LDS copy (ASYNCcnt-tracked), inline asm path ----
__device__ __forceinline__ void async_cp16(const _Float16* g, _Float16* l) {
  uint32_t loff = (uint32_t)(uintptr_t)l;   // flat LDS ptr low 32 bits = LDS offset
  asm volatile("global_load_async_to_lds_b128 %0, %1, off"
               :: "v"(loff), "v"(g) : "memory");
}
__device__ __forceinline__ void wait_async0() {
  asm volatile("s_wait_asynccnt 0x0" ::: "memory");
}

// Build A fragment (16x32 f16, MxK): lane holds row m=lane&15,
//   a[j] = row[8h+j], a[8+j] = row[16+8h+j]
__device__ __forceinline__ v16h mk_afrag(const _Float16* rb, int h) {
  v8h lo = *(const v8h*)(rb + 8*h);
  v8h hi = *(const v8h*)(rb + 16 + 8*h);
  v16h a;
#pragma unroll
  for (int j = 0; j < 8; ++j) { a[j] = lo[j]; a[8+j] = hi[j]; }
  return a;
}

__device__ __forceinline__ v16h mk_afrag_f32(const float* rb, int h) {
  v16h a;
#pragma unroll
  for (int j = 0; j < 8; ++j) {
    a[j]   = (_Float16)rb[8*h + j];
    a[8+j] = (_Float16)rb[16 + 8*h + j];
  }
  return a;
}

__device__ __forceinline__ float silu(float v) {
  return v * __builtin_amdgcn_rcpf(1.0f + __expf(-v));
}

// ---------------- weight swizzle: f32 row-major -> f16 B-fragment order ----
// B frag (32x16 KxN): lane l (h=l>>4, n=16*nt+(l&15)) holds b[i]=W[32kt+8h+i+(i>=8?8:0)][n]
__global__ __launch_bounds__(256) void swizzle_kernel(
    const float* __restrict__ mW1, const float* __restrict__ mW2,
    const float* __restrict__ uW1, const float* __restrict__ uW2,
    _Float16* __restrict__ out) {
  int idx = blockIdx.x * 256 + threadIdx.x;           // total DEPTH*52*512
  int L = idx / (FRAGS_PER_LAYER * 512);
  int r = idx % (FRAGS_PER_LAYER * 512);
  int frag = r >> 9;
  int t = r & 511;
  int lane = t >> 4, i = t & 15;
  const float* W; int Kreal, fb;
  if      (frag < 20) { W = mW1 + (size_t)L*140*64; Kreal = 140; fb = frag;      }
  else if (frag < 28) { W = mW2 + (size_t)L*64*64;  Kreal = 64;  fb = frag - 20; }
  else if (frag < 44) { W = uW1 + (size_t)L*128*64; Kreal = 128; fb = frag - 28; }
  else                { W = uW2 + (size_t)L*64*64;  Kreal = 64;  fb = frag - 44; }
  int kt = fb >> 2, nt = fb & 3;
  int K = 32*kt + 8*(lane >> 4) + i + ((i >= 8) ? 8 : 0);
  int n = 16*nt + (lane & 15);
  float v = (K < Kreal) ? W[(size_t)K*64 + n] : 0.0f;
  out[idx] = (_Float16)v;
}

// ---------------- embed: x = embed[z], xh = f16(x) ----------------
__global__ __launch_bounds__(256) void embed_kernel(
    const int* __restrict__ z, const float* __restrict__ embed,
    float* __restrict__ x, _Float16* __restrict__ xh) {
  int idx = blockIdx.x * 256 + threadIdx.x;   // NN*64
  int node = idx >> 6, col = idx & 63;
  float v = embed[z[node]*64 + col];
  x[idx] = v;
  xh[idx] = (_Float16)v;
}

// ---------------- Bessel RBF with Envelope(p=6), f16, padded to 16 -------
__global__ __launch_bounds__(256) void rbf_kernel(
    const int* __restrict__ ei, const float* __restrict__ pos,
    const float* __restrict__ freqs, _Float16* __restrict__ rbf) {
  int e = blockIdx.x * 256 + threadIdx.x;     // NE exact
  int s = ei[e], d = ei[NE + e];
  float dx = pos[3*s]   - pos[3*d];
  float dy = pos[3*s+1] - pos[3*d+1];
  float dz = pos[3*s+2] - pos[3*d+2];
  float dist = sqrtf(dx*dx + dy*dy + dz*dz);
  float t = dist * 0.25f;                     // dist / CUTOFF
  float t2 = t*t, t4 = t2*t2, t5 = t4*t, t6 = t5*t, t7 = t6*t;
  float env = __builtin_amdgcn_rcpf(t) - 28.0f*t5 + 48.0f*t6 - 21.0f*t7;
  if (!(t < 1.0f)) env = 0.0f;
  v8h o0, o1;
#pragma unroll
  for (int j = 0; j < 8; ++j) o0[j] = (_Float16)(env * sinf(freqs[j] * t));
#pragma unroll
  for (int j = 0; j < 4; ++j) o1[j] = (_Float16)(env * sinf(freqs[8+j] * t));
#pragma unroll
  for (int j = 4; j < 8; ++j) o1[j] = (_Float16)0.0f;
  *(v8h*)(rbf + (size_t)e*16)     = o0;
  *(v8h*)(rbf + (size_t)e*16 + 8) = o1;
}

// ------- fused edge MLP + scatter-add; persistent blocks, LDS weights -----
__global__ __launch_bounds__(256) void edge_kernel(
    const int* __restrict__ ei, const _Float16* __restrict__ xh,
    const _Float16* __restrict__ rbf, const _Float16* __restrict__ wl,
    const float* __restrict__ b1, const float* __restrict__ b2,
    float* __restrict__ agg) {
  __shared__ _Float16 Wlds[28 * 512];          // mW1s(20 frags) + mW2s(8 frags), 28 KB
  __shared__ _Float16 TileS[8 * 16 * TS];      // per-wave transpose tiles, 18 KB

  // async-stage this layer's weights into LDS: 1792 x 16B chunks / 256 thr = 7 each
#pragma unroll
  for (int i = 0; i < 7; ++i) {
    int c = threadIdx.x + i * 256;
    async_cp16(wl + (size_t)c * 8, &Wlds[c * 8]);
  }
  wait_async0();
  __syncthreads();

  const int lane = threadIdx.x & 31;
  const int wave = threadIdx.x >> 5;
  const int h = lane >> 4, m = lane & 15;
  _Float16* T = TileS + wave * 16 * TS;
  const _Float16* w1s = Wlds;
  const _Float16* w2s = Wlds + 20 * 512;

  for (int tile = blockIdx.x * 8 + wave; tile < NTILE_E; tile += EDGE_BLOCKS * 8) {
    const int e = tile * 16 + m;
    const int src = ei[e];
    const int dst = ei[NE + e];

    // A fragments for layer 1: [x_dst | x_src | rbf(pad32)], K = 160
    v16h A[5];
    {
      const _Float16* xd = xh + (size_t)dst * 64;
      const _Float16* xs = xh + (size_t)src * 64;
      A[0] = mk_afrag(xd,      h);
      A[1] = mk_afrag(xd + 32, h);
      A[2] = mk_afrag(xs,      h);
      A[3] = mk_afrag(xs + 32, h);
      v8h rl = *(const v8h*)(rbf + (size_t)e*16 + 8*h);
      v16h a4;
#pragma unroll
      for (int j = 0; j < 8; ++j) { a4[j] = rl[j]; a4[8+j] = (_Float16)0.0f; }
      A[4] = a4;
    }

    // layer 1: 20 WMMAs (B frags from LDS), SiLU, stage f16 into LDS tile
#pragma unroll
    for (int nt = 0; nt < 4; ++nt) {
      v8f c;
      float bv = b1[16*nt + m];
#pragma unroll
      for (int r = 0; r < 8; ++r) c[r] = bv;
#pragma unroll
      for (int kt = 0; kt < 5; ++kt) {
        v16h b = *(const v16h*)(w1s + (size_t)((kt*4 + nt)*32 + lane)*16);
        c = __builtin_amdgcn_wmma_f32_16x16x32_f16(false, A[kt], false, b,
                                                   (short)0, c, false, false);
      }
#pragma unroll
      for (int r = 0; r < 8; ++r)
        T[(8*h + r) * TS + 16*nt + m] = (_Float16)silu(c[r]);
    }

    // layer 2: 8 WMMAs, then f32 atomic scatter into agg
    int drow[8];
#pragma unroll
    for (int r = 0; r < 8; ++r) drow[r] = ei[NE + tile*16 + 8*h + r];

#pragma unroll
    for (int nt = 0; nt < 4; ++nt) {
      v8f c;
      float bv = b2[16*nt + m];
#pragma unroll
      for (int r = 0; r < 8; ++r) c[r] = bv;
#pragma unroll
      for (int kt = 0; kt < 2; ++kt) {
        v16h a2 = mk_afrag(T + m*TS + 32*kt, h);
        v16h b  = *(const v16h*)(w2s + (size_t)((kt*4 + nt)*32 + lane)*16);
        c = __builtin_amdgcn_wmma_f32_16x16x32_f16(false, a2, false, b,
                                                   (short)0, c, false, false);
      }
#pragma unroll
      for (int r = 0; r < 8; ++r)
        unsafeAtomicAdd(agg + (size_t)drow[r]*64 + 16*nt + m, c[r]);
    }
  }
}

// ------- fused node update MLP + residual; persistent, LDS weights --------
__global__ __launch_bounds__(256) void node_kernel(
    const float* __restrict__ agg, const _Float16* __restrict__ wl,
    const float* __restrict__ b1, const float* __restrict__ b2,
    float* __restrict__ x, _Float16* __restrict__ xh) {
  __shared__ _Float16 Wlds[24 * 512];          // uW1s(16) + uW2s(8), 24 KB
  __shared__ _Float16 TileS[8 * 16 * TS];

  // 1536 x 16B chunks / 256 threads = 6 each
#pragma unroll
  for (int i = 0; i < 6; ++i) {
    int c = threadIdx.x + i * 256;
    async_cp16(wl + (size_t)c * 8, &Wlds[c * 8]);
  }
  wait_async0();
  __syncthreads();

  const int lane = threadIdx.x & 31;
  const int wave = threadIdx.x >> 5;
  const int h = lane >> 4, m = lane & 15;
  _Float16* T = TileS + wave * 16 * TS;
  const _Float16* w1s = Wlds;
  const _Float16* w2s = Wlds + 16 * 512;

  for (int tile = blockIdx.x * 8 + wave; tile < NTILE_N; tile += NODE_BLOCKS * 8) {
    const int node = tile * 16 + m;

    // A for layer 1: [x | agg], K = 128
    v16h A[4];
    {
      const _Float16* xr = xh + (size_t)node * 64;
      const float*    ar = agg + (size_t)node * 64;
      A[0] = mk_afrag(xr,      h);
      A[1] = mk_afrag(xr + 32, h);
      A[2] = mk_afrag_f32(ar,      h);
      A[3] = mk_afrag_f32(ar + 32, h);
    }

#pragma unroll
    for (int nt = 0; nt < 4; ++nt) {
      v8f c;
      float bv = b1[16*nt + m];
#pragma unroll
      for (int r = 0; r < 8; ++r) c[r] = bv;
#pragma unroll
      for (int kt = 0; kt < 4; ++kt) {
        v16h b = *(const v16h*)(w1s + (size_t)((kt*4 + nt)*32 + lane)*16);
        c = __builtin_amdgcn_wmma_f32_16x16x32_f16(false, A[kt], false, b,
                                                   (short)0, c, false, false);
      }
#pragma unroll
      for (int r = 0; r < 8; ++r)
        T[(8*h + r) * TS + 16*nt + m] = (_Float16)silu(c[r]);
    }

#pragma unroll
    for (int nt = 0; nt < 4; ++nt) {
      v8f c;
      float bv = b2[16*nt + m];
#pragma unroll
      for (int r = 0; r < 8; ++r) c[r] = bv;
#pragma unroll
      for (int kt = 0; kt < 2; ++kt) {
        v16h a2 = mk_afrag(T + m*TS + 32*kt, h);
        v16h b  = *(const v16h*)(w2s + (size_t)((kt*4 + nt)*32 + lane)*16);
        c = __builtin_amdgcn_wmma_f32_16x16x32_f16(false, a2, false, b,
                                                   (short)0, c, false, false);
      }
#pragma unroll
      for (int r = 0; r < 8; ++r) {
        size_t off = (size_t)(tile*16 + 8*h + r) * 64 + 16*nt + m;
        float nx = x[off] + c[r];                  // residual
        x[off] = nx;
        xh[off] = (_Float16)nx;
      }
    }
  }
}

// ---------------- graph pooling (atomic segment sum) ----------------
__global__ __launch_bounds__(256) void pool_kernel(
    const int* __restrict__ batch, const float* __restrict__ x,
    float* __restrict__ pooled) {
  int idx = blockIdx.x * 256 + threadIdx.x;      // NN*64
  int node = idx >> 6, col = idx & 63;
  unsafeAtomicAdd(pooled + (size_t)batch[node]*64 + col, x[idx]);
}

// ---------------- readout head (tiny, plain VALU) ----------------
__global__ __launch_bounds__(256) void head_kernel(
    const float* __restrict__ pooled,
    const float* __restrict__ hW1, const float* __restrict__ hb1,
    const float* __restrict__ hW2, const float* __restrict__ hb2,
    float* __restrict__ out) {
  int g = blockIdx.x * 256 + threadIdx.x;
  if (g >= NG) return;
  float pg[64];
#pragma unroll
  for (int k = 0; k < 64; ++k) pg[k] = pooled[(size_t)g*64 + k];
  float o = hb2[0];
  for (int j = 0; j < 64; ++j) {
    float s = hb1[j];
#pragma unroll
    for (int k = 0; k < 64; ++k) s += pg[k] * hW1[k*64 + j];
    o += silu(s) * hW2[j];
  }
  out[g] = o;
}

// ---------------- launch ----------------
extern "C" void kernel_launch(void* const* d_in, const int* in_sizes, int n_in,
                              void* d_out, int out_size, void* d_ws, size_t ws_size,
                              hipStream_t stream) {
  const int*   z     = (const int*)  d_in[0];
  const float* pos   = (const float*)d_in[1];
  const int*   ei    = (const int*)  d_in[2];
  const int*   batch = (const int*)  d_in[3];
  const float* embed = (const float*)d_in[4];
  const float* freqs = (const float*)d_in[5];
  const float* mW1   = (const float*)d_in[6];
  const float* mb1   = (const float*)d_in[7];
  const float* mW2   = (const float*)d_in[8];
  const float* mb2   = (const float*)d_in[9];
  const float* uW1   = (const float*)d_in[10];
  const float* ub1   = (const float*)d_in[11];
  const float* uW2   = (const float*)d_in[12];
  const float* ub2   = (const float*)d_in[13];
  const float* hW1   = (const float*)d_in[14];
  const float* hb1   = (const float*)d_in[15];
  const float* hW2   = (const float*)d_in[16];
  const float* hb2   = (const float*)d_in[17];

  char* ws = (char*)d_ws;
  float*    x      = (float*)   (ws + OFF_X);
  float*    agg    = (float*)   (ws + OFF_AGG);
  _Float16* xh     = (_Float16*)(ws + OFF_XH);
  _Float16* rbf    = (_Float16*)(ws + OFF_RBF);
  float*    pooled = (float*)   (ws + OFF_POOL);
  _Float16* wq     = (_Float16*)(ws + OFF_W);

  // one-time (per call) prep: weight swizzle, embedding, RBF
  swizzle_kernel<<<(DEPTH * FRAGS_PER_LAYER * 512) / 256, 256, 0, stream>>>(
      mW1, mW2, uW1, uW2, wq);
  embed_kernel<<<(NN * 64) / 256, 256, 0, stream>>>(z, embed, x, xh);
  rbf_kernel<<<NE / 256, 256, 0, stream>>>(ei, pos, freqs, rbf);

  for (int l = 0; l < DEPTH; ++l) {
    hipMemsetAsync(agg, 0, (size_t)NN * 64 * 4, stream);
    const _Float16* wl = wq + (size_t)l * WLAYER;
    edge_kernel<<<EDGE_BLOCKS, 256, 0, stream>>>(
        ei, xh, rbf, wl, mb1 + l * 64, mb2 + l * 64, agg);
    node_kernel<<<NODE_BLOCKS, 256, 0, stream>>>(
        agg, wl + 28 * 512, ub1 + l * 64, ub2 + l * 64, x, xh);
  }

  hipMemsetAsync(pooled, 0, (size_t)NG * 64 * 4, stream);
  pool_kernel<<<(NN * 64) / 256, 256, 0, stream>>>(batch, x, pooled);
  head_kernel<<<(NG + 255) / 256, 256, 0, stream>>>(pooled, hW1, hb1, hW2, hb2,
                                                    (float*)d_out);
}
